// MyAttention_24790551232728
// MI455X (gfx1250) — matrix operand order repaired and checked by reading, not compiled, not run on hardware
//
#include <hip/hip_runtime.h>

// ---------------- types ----------------
typedef __attribute__((ext_vector_type(16))) _Float16 v16h;
typedef __attribute__((ext_vector_type(8)))  _Float16 v8h;
typedef __attribute__((ext_vector_type(8)))  float    v8f;

// ---------------- problem constants ----------------
#define BDIM   2
#define TDIM   4096
#define DDIM   768
#define DODIM  256
#define W1C    7
#define W2C    15
#define NPSEQ  4110        // T + 2*W1 (padded sequence length)
#define NPAD   8220        // B * NPSEQ  (rows that get the bias)
#define RPOS   8220        // first pos_emb row in packed X
#define RTOT   8256        // 516 row tiles of 16; rows >= 8235 are hard zeros

// ---------------- WMMA operand loaders (wave32, 16x16x32 f16) ----------------
// 16 contiguous f16 -> v16h (two 16B loads)
__device__ inline v16h ld16c(const _Float16* p) {
  v8h lo = *(const v8h*)p;
  v8h hi = *(const v8h*)(p + 8);
  return __builtin_shufflevector(lo, hi, 0,1,2,3,4,5,6,7,8,9,10,11,12,13,14,15);
}

// A-matrix 16x32 f16, row-major with leading dim `ld`.
// ISA layout: lanes 0-15 hold row M=lane, K = 0..7 then 16..23;
//             lanes 16-31 hold row M=lane-16, K = 8..15 then 24..31.
__device__ inline v16h load_A16x32(const _Float16* tile, int ld, int lane) {
  int m = lane & 15;
  int s = (lane >> 4) * 8;
  const _Float16* row = tile + (size_t)m * ld + s;
  v8h a0 = *(const v8h*)row;         // K = s .. s+7
  v8h a1 = *(const v8h*)(row + 16);  // K = 16+s .. 16+s+7
  return __builtin_shufflevector(a0, a1, 0,1,2,3,4,5,6,7,8,9,10,11,12,13,14,15);
}

// B-matrix 32x16 f16 where this lane's column is contiguous in K.
// p = pointer to element (k=0) of this lane's column (caller adds column row).
// ISA layout: lanes 0-15 (N=lane) hold K=0..7,16..23; lanes 16-31 hold K=8..15,24..31.
__device__ inline v16h load_Bcontig(const _Float16* p, int lane) {
  const _Float16* q = p + ((lane >> 4) << 3);
  v8h lo = *(const v8h*)q;
  v8h hi = *(const v8h*)(q + 16);
  return __builtin_shufflevector(lo, hi, 0,1,2,3,4,5,6,7,8,9,10,11,12,13,14,15);
}

// B-matrix 32x16 f16 where K advances with stride `ld` in memory.
__device__ inline v16h load_Bstride(const _Float16* p, int ld, int lane) {
  int kh = (lane >> 4) << 3;
  v16h b = {};
#pragma unroll
  for (int j = 0; j < 16; ++j) b[j] = p[(size_t)(kh + (j & 7) + ((j >> 3) << 4)) * ld];
  return b;
}

__device__ inline v8f wmma_f16(v16h a, v16h b, v8f c) {
  return __builtin_amdgcn_wmma_f32_16x16x32_f16(false, a, false, b, (short)0, c,
                                                false, false);
}

// ---------------- kernel 0: pack padded inputs + pos_emb -> f16 X ----------------
__global__ void lw_pack_x(const float* __restrict__ inputs,
                          const float* __restrict__ pos_emb,
                          _Float16* __restrict__ Xh) {
  size_t idx = (size_t)blockIdx.x * blockDim.x + threadIdx.x;
  if (idx >= (size_t)RTOT * DDIM) return;
  int r = (int)(idx / DDIM);
  int k = (int)(idx % DDIM);
  float v = 0.f;
  if (r < NPAD) {
    int b = r / NPSEQ;
    int t = (r % NPSEQ) - W1C;
    if (t >= 0 && t < TDIM) v = inputs[((size_t)b * TDIM + t) * DDIM + k];
  } else if (r < RPOS + W2C) {
    v = pos_emb[(size_t)(r - RPOS) * DDIM + k];
  }
  Xh[idx] = (_Float16)v;
}

// ---------------- kernel 1: pack weights -> f16 ----------------
__global__ void lw_pack_w(const float* __restrict__ Wq, const float* __restrict__ Wv,
                          _Float16* __restrict__ Wqh, _Float16* __restrict__ Wvh) {
  size_t idx = (size_t)blockIdx.x * blockDim.x + threadIdx.x;
  if (idx >= (size_t)DODIM * DDIM) return;
  Wqh[idx] = (_Float16)Wq[idx];
  Wvh[idx] = (_Float16)Wv[idx];
}

// ---------------- kernel 2: fused Q/V projection GEMM (WMMA f16) ----------------
// out[m,n] = sum_k X[m,k] * W[n,k] (+ bias for real rows); stored as f16.
__global__ void lw_proj(const _Float16* __restrict__ Xh,
                        const _Float16* __restrict__ Wqh,
                        const _Float16* __restrict__ Wvh,
                        const float* __restrict__ bq, const float* __restrict__ bv,
                        _Float16* __restrict__ QP, _Float16* __restrict__ VP) {
  int lane = threadIdx.x;           // one wave per block
  int ct = blockIdx.x;              // 0..15  (DO/16)
  int rt = blockIdx.y;              // 0..RTOT/16-1
  int n = lane & 15;
  v8f accq = {}; v8f accv = {};
  const _Float16* xtile = Xh + (size_t)rt * 16 * DDIM;
  const _Float16* wqcol = Wqh + (size_t)(ct * 16 + n) * DDIM;  // B[k][n] = W[n][k]
  const _Float16* wvcol = Wvh + (size_t)(ct * 16 + n) * DDIM;
#pragma unroll 4
  for (int k0 = 0; k0 < DDIM; k0 += 32) {
    v16h a = load_A16x32(xtile + k0, DDIM, lane);
    accq = wmma_f16(a, load_Bcontig(wqcol + k0, lane), accq);
    accv = wmma_f16(a, load_Bcontig(wvcol + k0, lane), accv);
  }
  int col = ct * 16 + n;
  int ro = (lane >> 4) * 8;
  float bqv = bq[col], bvv = bv[col];
#pragma unroll
  for (int r = 0; r < 8; ++r) {
    int grow = rt * 16 + ro + r;
    float aq = (grow < NPAD) ? bqv : 0.f;   // pos_emb rows and zero tail: no bias
    float av = (grow < NPAD) ? bvv : 0.f;
    QP[(size_t)grow * DODIM + col] = (_Float16)(accq[r] + aq);
    VP[(size_t)grow * DODIM + col] = (_Float16)(accv[r] + av);
  }
}

// ---------------- kernel 3: banded windowed attention (WMMA f16) ----------------
// grid: (T/16, H, B), block 32 (one wave).
__global__ void lw_attn(const _Float16* __restrict__ QP,
                        const _Float16* __restrict__ VP,
                        float* __restrict__ out) {
  __shared__ float sfull[16][32];   // Q . Kwin^T, cols s = 0..29 (+2 junk)
  __shared__ float pmat[16][16];    // Q . pos_q^T
  __shared__ float sc[16][16];      // scores, col 15 = 0
  int lane = threadIdx.x;
  int t0 = blockIdx.x * 16;
  int h  = blockIdx.y;
  int b  = blockIdx.z;
  int rowBase = b * NPSEQ;
  int colBase = h * 64;
  int n = lane & 15;
  int ro = (lane >> 4) * 8;

  // ---- scores: Sfull[t,s] = q[t] . q_pad[t0+s],  P[t,w] = q[t] . pos_q[w]
  v8f accS0 = {}; v8f accS1 = {}; v8f accP = {};
  const _Float16* qtile = QP + (size_t)(rowBase + W1C + t0) * DODIM + colBase;
  const _Float16* krow0 = QP + (size_t)(rowBase + t0 + n) * DODIM + colBase;
  const _Float16* krow1 = QP + (size_t)(rowBase + t0 + 16 + n) * DODIM + colBase;
  const _Float16* prow  = QP + (size_t)(RPOS + n) * DODIM + colBase;   // row 15 is zero
#pragma unroll
  for (int k0 = 0; k0 < 64; k0 += 32) {
    v16h a = load_A16x32(qtile + k0, DODIM, lane);
    accS0 = wmma_f16(a, load_Bcontig(krow0 + k0, lane), accS0);
    accS1 = wmma_f16(a, load_Bcontig(krow1 + k0, lane), accS1);
    accP  = wmma_f16(a, load_Bcontig(prow  + k0, lane), accP);
  }
#pragma unroll
  for (int r = 0; r < 8; ++r) {
    sfull[ro + r][n]      = accS0[r];
    sfull[ro + r][16 + n] = accS1[r];
    pmat [ro + r][n]      = accP[r];
  }
  __syncthreads();
  // band gather: scores[i][w] = Sfull[i][i+w] + P[i][w]
  {
    int i = n, wb = (lane >> 4) * 8;
#pragma unroll
    for (int dw = 0; dw < 8; ++dw) {
      int w = wb + dw;
      sc[i][w] = (w < W2C) ? (sfull[i][i + w] + pmat[i][w]) : 0.f;
    }
  }
  __syncthreads();

  // ---- context: ctx = band(scores) * Vwin  +  scores * pos_v
  int m = n, s8 = (lane >> 4) * 8;
  v16h aS = {}, aB = {};
#pragma unroll
  for (int j = 0; j < 8; ++j) {
    int k1 = s8 + j, k2 = 16 + s8 + j;
    aS[j]     = (_Float16)((k1 < W2C) ? sc[m][k1] : 0.f);
    aS[8 + j] = (_Float16)((k2 < W2C) ? sc[m][k2] : 0.f);
    aB[j]     = (_Float16)((k1 >= m && k1 - m < W2C) ? sc[m][k1 - m] : 0.f);
    aB[8 + j] = (_Float16)((k2 >= m && k2 - m < W2C) ? sc[m][k2 - m] : 0.f);
  }
#pragma unroll
  for (int c = 0; c < 4; ++c) {
    v8f acc = {};
    const _Float16* pv = VP + (size_t)RPOS * DODIM + colBase + c * 16 + n;            // rows >=15 zero
    const _Float16* vw = VP + (size_t)(rowBase + t0) * DODIM + colBase + c * 16 + n;  // k>=30 killed by aB
    acc = wmma_f16(aS, load_Bstride(pv, DODIM, lane), acc);
    acc = wmma_f16(aB, load_Bstride(vw, DODIM, lane), acc);
#pragma unroll
    for (int r = 0; r < 8; ++r) {
      int t = t0 + ro + r;
      out[((size_t)(b * TDIM + t)) * DODIM + colBase + c * 16 + n] = acc[r];
    }
  }
}

// ---------------- launcher ----------------
extern "C" void kernel_launch(void* const* d_in, const int* in_sizes, int n_in,
                              void* d_out, int out_size, void* d_ws, size_t ws_size,
                              hipStream_t stream) {
  (void)in_sizes; (void)n_in; (void)out_size; (void)ws_size;
  const float* inputs  = (const float*)d_in[0];
  const float* Wq      = (const float*)d_in[1];
  const float* bq      = (const float*)d_in[2];
  const float* Wv      = (const float*)d_in[3];
  const float* bv      = (const float*)d_in[4];
  const float* pos_emb = (const float*)d_in[5];
  float* out = (float*)d_out;

  // workspace layout (f16, all 16B-aligned): X | Wq | Wv | QP | VP  (~22 MB)
  _Float16* Xh  = (_Float16*)d_ws;
  _Float16* Wqh = Xh  + (size_t)RTOT * DDIM;
  _Float16* Wvh = Wqh + (size_t)DODIM * DDIM;
  _Float16* QP  = Wvh + (size_t)DODIM * DDIM;
  _Float16* VP  = QP  + (size_t)RTOT * DODIM;

  {
    size_t tot = (size_t)RTOT * DDIM;
    lw_pack_x<<<(unsigned)((tot + 255) / 256), 256, 0, stream>>>(inputs, pos_emb, Xh);
  }
  {
    size_t tot = (size_t)DODIM * DDIM;
    lw_pack_w<<<(unsigned)((tot + 255) / 256), 256, 0, stream>>>(Wq, Wv, Wqh, Wvh);
  }
  lw_proj<<<dim3(DODIM / 16, RTOT / 16), 32, 0, stream>>>(Xh, Wqh, Wvh, bq, bv, QP, VP);
  lw_attn<<<dim3(TDIM / 16, 4, BDIM), 32, 0, stream>>>(QP, VP, out);
}
